// TransfomerBackbone_80513456931525
// MI455X (gfx1250) — matrix operand order repaired and checked by reading, not compiled, not run on hardware
//
#include <hip/hip_runtime.h>
#include <hip/hip_bf16.h>
#include <math.h>

typedef __attribute__((ext_vector_type(16))) __bf16 v16bf;
typedef __attribute__((ext_vector_type(8)))  __bf16 v8bf;
typedef __attribute__((ext_vector_type(4)))  __bf16 v4bf;
typedef __attribute__((ext_vector_type(8)))  float  v8f;
typedef __attribute__((ext_vector_type(4)))  float  v4f;

#define K_NN 16
#define HID  512

// ------------------------------------------------------------------
// WMMA GEMM: C[M,N] = epilogue( A[M,K] @ B[K,N] )
// Block tile 128x64, BK=32, 8 waves; each wave computes 32x32 via
// 4x v_wmma_f32_16x16x32_bf16 (2 A-frags x 2 B-frags).
// A stored row-major in LDS (pitch 40), B stored TRANSPOSED (pitch 40)
// so each lane's fragment is two contiguous 16B runs -> ds_load_b128.
// Fast path (interior tiles, K%4==0 / N%4==0): float4 global loads.
// ------------------------------------------------------------------
#define TBM 128
#define TBN 64
#define TBK 32
#define LDA 40
#define LDB 40

__device__ __forceinline__ void epi_store(float* C, const float* bias,
                                          const float* resid, int relu,
                                          int M, int N, int r, int cc, float x)
{
  if (r < M && cc < N) {
    if (bias)  x += bias[cc];
    if (resid) x += resid[(size_t)r * N + cc];
    if (relu)  x = fmaxf(x, 0.0f);
    C[(size_t)r * N + cc] = x;
  }
}

__global__ __launch_bounds__(256)
void k_gemm(const float* __restrict__ A, const float* __restrict__ B,
            const float* __restrict__ bias, const float* resid,
            float* C, int M, int N, int K, int relu)
{
  __shared__ __bf16 sA[TBM][LDA];
  __shared__ __bf16 sBt[TBN][LDB];   // [col][k]
  const int tid  = threadIdx.x;
  const int lane = tid & 31;
  const int wave = tid >> 5;
  const int wm   = (wave >> 1) * 32;   // 0,32,64,96
  const int wn   = (wave & 1) * 32;    // 0,32
  const int bm0  = blockIdx.y * TBM;
  const int bn0  = blockIdx.x * TBN;

  v8f acc00 = {}, acc01 = {}, acc10 = {}, acc11 = {};

  const bool mfull = (bm0 + TBM <= M);
  const bool nfull = (bn0 + TBN <= N);
  const bool k4 = ((K & 3) == 0);
  const bool n4 = ((N & 3) == 0);

  for (int k0 = 0; k0 < K; k0 += TBK) {
    const bool kfull = (k0 + TBK <= K);

    if (mfull && kfull && k4) {           // fast A: 4 x float4 per thread
#pragma unroll
      for (int i = 0; i < 4; ++i) {
        int g = i * 256 + tid;
        int r = g >> 3, c4 = (g & 7) << 2;
        const v4f av = *(const v4f*)(A + (size_t)(bm0 + r) * K + k0 + c4);
        v4bf bv;
        bv[0] = (__bf16)av[0]; bv[1] = (__bf16)av[1];
        bv[2] = (__bf16)av[2]; bv[3] = (__bf16)av[3];
        *(v4bf*)&sA[r][c4] = bv;          // ds_store_b64
      }
    } else {                              // slow A: clamped addr + select
#pragma unroll
      for (int i = 0; i < 16; ++i) {
        int e = i * 256 + tid;
        int r = e >> 5, c = e & 31;
        int gr = bm0 + r, gc = k0 + c;
        int cr = gr < M ? gr : M - 1;
        int cc = gc < K ? gc : K - 1;
        float v = A[(size_t)cr * K + cc];
        if (gr >= M || gc >= K) v = 0.0f;
        sA[r][c] = (__bf16)v;
      }
    }
    if (nfull && kfull && n4) {           // fast B: 2 x float4, transpose store
#pragma unroll
      for (int i = 0; i < 2; ++i) {
        int g = i * 256 + tid;
        int r = g >> 4, c4 = (g & 15) << 2;
        const v4f bv = *(const v4f*)(B + (size_t)(k0 + r) * N + bn0 + c4);
#pragma unroll
        for (int j = 0; j < 4; ++j) sBt[c4 + j][r] = (__bf16)bv[j];
      }
    } else {
#pragma unroll
      for (int i = 0; i < 8; ++i) {
        int e = i * 256 + tid;
        int r = e >> 6, c = e & 63;
        int gr = k0 + r, gc = bn0 + c;
        int cr = gr < K ? gr : K - 1;
        int cc = gc < N ? gc : N - 1;
        float v = B[(size_t)cr * N + cc];
        if (gr >= K || gc >= N) v = 0.0f;
        sBt[c][r] = (__bf16)v;
      }
    }
    __syncthreads();

    // A fragment (ISA 16-bit A 16x32): lane<16 row=lane K{0..7,16..23},
    // lane>=16 row=lane-16 K{8..15,24..31} -> two contiguous 16B runs.
    const int rl = lane & 15;
    const int kh = (lane >> 4) * 8;
    v8bf a0lo = *(const v8bf*)&sA[wm + rl][kh];
    v8bf a0hi = *(const v8bf*)&sA[wm + rl][kh + 16];
    v8bf a1lo = *(const v8bf*)&sA[wm + 16 + rl][kh];
    v8bf a1hi = *(const v8bf*)&sA[wm + 16 + rl][kh + 16];
    v16bf a0 = __builtin_shufflevector(a0lo, a0hi, 0,1,2,3,4,5,6,7,8,9,10,11,12,13,14,15);
    v16bf a1 = __builtin_shufflevector(a1lo, a1hi, 0,1,2,3,4,5,6,7,8,9,10,11,12,13,14,15);
    // B fragment: col = lane&15 (+16 for hi tile), K = kb..kb+7, kb+16..kb+23 (same as A)
    const int kb = (lane >> 4) * 8;
    v8bf b0lo = *(const v8bf*)&sBt[wn + rl][kb];
    v8bf b0hi = *(const v8bf*)&sBt[wn + rl][kb + 16];
    v8bf b1lo = *(const v8bf*)&sBt[wn + 16 + rl][kb];
    v8bf b1hi = *(const v8bf*)&sBt[wn + 16 + rl][kb + 16];
    v16bf b0 = __builtin_shufflevector(b0lo, b0hi, 0,1,2,3,4,5,6,7,8,9,10,11,12,13,14,15);
    v16bf b1 = __builtin_shufflevector(b1lo, b1hi, 0,1,2,3,4,5,6,7,8,9,10,11,12,13,14,15);

    acc00 = __builtin_amdgcn_wmma_f32_16x16x32_bf16(false, a0, false, b0, (short)0, acc00, false, false);
    acc01 = __builtin_amdgcn_wmma_f32_16x16x32_bf16(false, a0, false, b1, (short)0, acc01, false, false);
    acc10 = __builtin_amdgcn_wmma_f32_16x16x32_bf16(false, a1, false, b0, (short)0, acc10, false, false);
    acc11 = __builtin_amdgcn_wmma_f32_16x16x32_bf16(false, a1, false, b1, (short)0, acc11, false, false);
    __syncthreads();
  }

  // C/D layout: vgpr v, lane -> row = v + 8*(lane>=16), col = lane&15
  const int cl = lane & 15;
  const int rh = (lane >> 4) * 8;
#pragma unroll
  for (int v = 0; v < 8; ++v) {
    int r0 = bm0 + wm + rh + v;
    int r1 = r0 + 16;
    int c0 = bn0 + wn + cl;
    int c1 = c0 + 16;
    epi_store(C, bias, resid, relu, M, N, r0, c0, acc00[v]);
    epi_store(C, bias, resid, relu, M, N, r0, c1, acc01[v]);
    epi_store(C, bias, resid, relu, M, N, r1, c0, acc10[v]);
    epi_store(C, bias, resid, relu, M, N, r1, c1, acc11[v]);
  }
}

// zero-pad a weight matrix along K (rows): out[rows_pad, cols]
__global__ void k_padw(const float* __restrict__ w, float* __restrict__ out,
                       int rows, int rows_pad, int cols)
{
  size_t t = (size_t)blockIdx.x * blockDim.x + threadIdx.x;
  size_t total = (size_t)rows_pad * cols;
  if (t >= total) return;
  int r = (int)(t / cols);
  out[t] = (r < rows) ? w[t] : 0.0f;
}

// ------------------------------------------------------------------
// Brute-force KNN with LDS-cached reference points (N<=4096)
// ------------------------------------------------------------------
__global__ __launch_bounds__(256)
void k_knn(const float* __restrict__ q, int M,
           const float* __restrict__ ref, int N,
           int* __restrict__ idx)
{
  __shared__ float srf[4096 * 3];
  for (int i = threadIdx.x; i < N * 3; i += blockDim.x) srf[i] = ref[i];
  __syncthreads();
  int m = blockIdx.x * blockDim.x + threadIdx.x;
  if (m >= M) return;
  float qx = q[m * 3 + 0], qy = q[m * 3 + 1], qz = q[m * 3 + 2];
  float bd[K_NN]; int bi[K_NN];
#pragma unroll
  for (int j = 0; j < K_NN; ++j) { bd[j] = 3.4e38f; bi[j] = 0; }
  for (int n = 0; n < N; ++n) {
    float dx = qx - srf[n * 3 + 0];
    float dy = qy - srf[n * 3 + 1];
    float dz = qz - srf[n * 3 + 2];
    float d = dx * dx + dy * dy + dz * dz;
    if (d < bd[K_NN - 1]) {
      bd[K_NN - 1] = d; bi[K_NN - 1] = n;
#pragma unroll
      for (int j = K_NN - 1; j > 0; --j) {
        if (bd[j] < bd[j - 1]) {
          float td = bd[j]; bd[j] = bd[j - 1]; bd[j - 1] = td;
          int   ti = bi[j]; bi[j] = bi[j - 1]; bi[j - 1] = ti;
        }
      }
    }
  }
#pragma unroll
  for (int j = 0; j < K_NN; ++j) idx[(size_t)m * K_NN + j] = bi[j];
}

// ------------------------------------------------------------------
// Farthest point sampling: single workgroup, LDS min-dist + argmax
// ------------------------------------------------------------------
__global__ __launch_bounds__(1024)
void k_fps(const float* __restrict__ xyz, int N, int S, float* __restrict__ out)
{
  __shared__ float mind[4096];
  __shared__ float rv[1024];
  __shared__ int   ri[1024];
  const int t = threadIdx.x;
  const int nt = blockDim.x;
  for (int i = t; i < N; i += nt) mind[i] = 3.4e38f;
  if (t == 0) { out[0] = xyz[0]; out[1] = xyz[1]; out[2] = xyz[2]; }
  int last = 0;
  __syncthreads();
  for (int s = 1; s < S; ++s) {
    float lx = xyz[last * 3 + 0], ly = xyz[last * 3 + 1], lz = xyz[last * 3 + 2];
    float bv = -1.0f; int bx = 0;
    for (int i = t; i < N; i += nt) {
      float dx = xyz[i * 3 + 0] - lx, dy = xyz[i * 3 + 1] - ly, dz = xyz[i * 3 + 2] - lz;
      float d = dx * dx + dy * dy + dz * dz;
      float mn = fminf(mind[i], d);
      mind[i] = mn;
      if (mn > bv) { bv = mn; bx = i; }
    }
    rv[t] = bv; ri[t] = bx;
    __syncthreads();
    for (int off = nt >> 1; off > 0; off >>= 1) {
      if (t < off && rv[t + off] > rv[t]) { rv[t] = rv[t + off]; ri[t] = ri[t + off]; }
      __syncthreads();
    }
    last = ri[0];
    if (t == 0) {
      out[s * 3 + 0] = xyz[last * 3 + 0];
      out[s * 3 + 1] = xyz[last * 3 + 1];
      out[s * 3 + 2] = xyz[last * 3 + 2];
    }
    __syncthreads();
  }
}

// ------------------------------------------------------------------
// Elementwise / gather kernels
// ------------------------------------------------------------------
__global__ void k_slice3(const float* __restrict__ ev, float* __restrict__ xyz, int N)
{
  int i = blockIdx.x * blockDim.x + threadIdx.x;
  if (i >= N) return;
  xyz[i * 3 + 0] = ev[i * 4 + 0];
  xyz[i * 3 + 1] = ev[i * 4 + 1];
  xyz[i * 3 + 2] = ev[i * 4 + 2];
}

__global__ void k_rel(const float* __restrict__ q, const float* __restrict__ ref,
                      const int* __restrict__ idx, float* __restrict__ out,
                      int M, int ostride, int ooff, float sign)
{
  int t = blockIdx.x * blockDim.x + threadIdx.x;
  if (t >= M * K_NN) return;
  int m = t >> 4;
  int nb = idx[t];
#pragma unroll
  for (int j = 0; j < 3; ++j)
    out[(size_t)t * ostride + ooff + j] = sign * (q[m * 3 + j] - ref[nb * 3 + j]);
}

__global__ void k_gatherf(const float* __restrict__ f, const int* __restrict__ idx,
                          float* __restrict__ out, int M, int F, int ostride, int ooff)
{
  size_t t = (size_t)blockIdx.x * blockDim.x + threadIdx.x;
  size_t total = (size_t)M * K_NN * F;
  if (t >= total) return;
  int c = (int)(t % F);
  size_t mk = t / F;
  int nb = idx[mk];
  out[mk * ostride + ooff + c] = f[(size_t)nb * F + c];
}

__global__ void k_qkpos(const float* __restrict__ q, const int* __restrict__ idx,
                        const float* __restrict__ kf, const float* __restrict__ pos,
                        float* __restrict__ out, int Mc)
{
  size_t t = (size_t)blockIdx.x * blockDim.x + threadIdx.x;
  size_t total = (size_t)Mc * K_NN * HID;
  if (t >= total) return;
  int h = (int)(t & (HID - 1));
  size_t mk = t / HID;
  int m = (int)(mk >> 4);
  int nb = idx[mk];
  out[t] = q[(size_t)m * HID + h] - kf[(size_t)nb * HID + h] + pos[t];
}

__global__ void k_softmax16(float* __restrict__ att, int Mc)
{
  size_t t = (size_t)blockIdx.x * blockDim.x + threadIdx.x;
  size_t total = (size_t)Mc * HID;
  if (t >= total) return;
  int h = (int)(t & (HID - 1));
  int m = (int)(t / HID);
  const float sc = 0.04419417382415922f; // 1/sqrt(512)
  size_t base = (size_t)m * K_NN * HID + h;
  float mx = -3.4e38f;
#pragma unroll
  for (int k = 0; k < K_NN; ++k) mx = fmaxf(mx, att[base + (size_t)k * HID] * sc);
  float e[K_NN]; float sum = 0.0f;
#pragma unroll
  for (int k = 0; k < K_NN; ++k) {
    e[k] = __expf(att[base + (size_t)k * HID] * sc - mx);
    sum += e[k];
  }
  float inv = 1.0f / sum;
#pragma unroll
  for (int k = 0; k < K_NN; ++k) att[base + (size_t)k * HID] = e[k] * inv;
}

__global__ void k_attnred(const float* __restrict__ att, const int* __restrict__ idx,
                          const float* __restrict__ vf, const float* __restrict__ pos,
                          float* __restrict__ res, int Mc)
{
  size_t t = (size_t)blockIdx.x * blockDim.x + threadIdx.x;
  size_t total = (size_t)Mc * HID;
  if (t >= total) return;
  int h = (int)(t & (HID - 1));
  int m = (int)(t / HID);
  size_t base = (size_t)m * K_NN * HID + h;
  float s = 0.0f;
#pragma unroll
  for (int k = 0; k < K_NN; ++k) {
    int nb = idx[(size_t)m * K_NN + k];
    s += att[base + (size_t)k * HID] * (vf[(size_t)nb * HID + h] + pos[base + (size_t)k * HID]);
  }
  res[t] = s;
}

__global__ void k_bnrelu(float* __restrict__ h, const float* __restrict__ mean,
                         const float* __restrict__ var, const float* __restrict__ gamma,
                         const float* __restrict__ beta, size_t total, int C)
{
  size_t t = (size_t)blockIdx.x * blockDim.x + threadIdx.x;
  if (t >= total) return;
  int c = (int)(t % C);
  float x = (h[t] - mean[c]) * rsqrtf(var[c] + 1e-5f) * gamma[c] + beta[c];
  h[t] = fmaxf(x, 0.0f);
}

__global__ void k_maxpool16(const float* __restrict__ in, float* __restrict__ out,
                            int M, int C)
{
  size_t t = (size_t)blockIdx.x * blockDim.x + threadIdx.x;
  size_t total = (size_t)M * C;
  if (t >= total) return;
  int c = (int)(t % C);
  int m = (int)(t / C);
  float mx = -3.4e38f;
#pragma unroll
  for (int k = 0; k < K_NN; ++k)
    mx = fmaxf(mx, in[((size_t)m * K_NN + k) * C + c]);
  out[t] = mx;
}

// ------------------------------------------------------------------
// Host orchestration
// ------------------------------------------------------------------
struct Lin { const float *b, *w; };
struct BNL { const float *b, *beta, *gamma, *mean, *var, *w; };
struct TBp {
  Lin agg, att0, att1, ker, pos0, pos1;
  const float *wk, *wq, *wv;
  const float *pos0w_pad;   // 4 x 512, zero-padded row 3
};

static inline size_t a256(size_t x) { return (x + 255) & ~(size_t)255; }
static inline int cdiv(int a, int b) { return (a + b - 1) / b; }

static void run_tb(const TBp& p, const float* xyz, const float* f_in, float* f_out,
                   int N, int F, char* ws, hipStream_t stream)
{
  char* c = ws;
  int*   idx = (int*)c;   c += a256((size_t)N * K_NN * 4);
  float* x   = (float*)c; c += a256((size_t)N * HID * 4);
  float* q   = (float*)c; c += a256((size_t)N * HID * 4);
  float* kf  = (float*)c; c += a256((size_t)N * HID * 4);
  float* vf  = (float*)c; c += a256((size_t)N * HID * 4);
  float* res = (float*)c; c += a256((size_t)N * HID * 4);
  const int CH = (N > 1024) ? 1024 : N;   // chunk the per-neighbor MLPs
  float* rel = (float*)c; c += a256((size_t)CH * K_NN * 4 * 4);  // K padded to 4
  float* P1  = (float*)c; c += a256((size_t)CH * K_NN * HID * 4);
  float* P2  = (float*)c; c += a256((size_t)CH * K_NN * HID * 4);
  float* A1  = (float*)c; c += a256((size_t)CH * K_NN * HID * 4);

  k_knn<<<cdiv(N, 256), 256, 0, stream>>>(xyz, N, xyz, N, idx);

  dim3 gx(cdiv(HID, TBN), cdiv(N, TBM));
  k_gemm<<<gx, 256, 0, stream>>>(f_in, p.ker.w, p.ker.b, nullptr, x, N, HID, F, 0);
  k_gemm<<<gx, 256, 0, stream>>>(x, p.wq, nullptr, nullptr, q,  N, HID, HID, 0);
  k_gemm<<<gx, 256, 0, stream>>>(x, p.wk, nullptr, nullptr, kf, N, HID, HID, 0);
  k_gemm<<<gx, 256, 0, stream>>>(x, p.wv, nullptr, nullptr, vf, N, HID, HID, 0);

  for (int m0 = 0; m0 < N; m0 += CH) {
    int Mc = (N - m0 < CH) ? (N - m0) : CH;
    const int* idxc = idx + (size_t)m0 * K_NN;
    // rel with row stride 4; pad column multiplies a zero weight row
    k_rel<<<cdiv(Mc * K_NN, 256), 256, 0, stream>>>(xyz + (size_t)m0 * 3, xyz, idxc,
                                                    rel, Mc, 4, 0, 1.0f);
    dim3 gp(cdiv(HID, TBN), cdiv(Mc * K_NN, TBM));
    k_gemm<<<gp, 256, 0, stream>>>(rel, p.pos0w_pad, p.pos0.b, nullptr, P1, Mc * K_NN, HID, 4, 1);
    k_gemm<<<gp, 256, 0, stream>>>(P1, p.pos1.w, p.pos1.b, nullptr, P2, Mc * K_NN, HID, HID, 0);

    size_t tot = (size_t)Mc * K_NN * HID;
    k_qkpos<<<(int)((tot + 255) / 256), 256, 0, stream>>>(q + (size_t)m0 * HID, idxc, kf, P2, A1, Mc);
    k_gemm<<<gp, 256, 0, stream>>>(A1, p.att0.w, p.att0.b, nullptr, P1, Mc * K_NN, HID, HID, 1);
    k_gemm<<<gp, 256, 0, stream>>>(P1, p.att1.w, p.att1.b, nullptr, A1, Mc * K_NN, HID, HID, 0);

    size_t tmh = (size_t)Mc * HID;
    k_softmax16<<<(int)((tmh + 255) / 256), 256, 0, stream>>>(A1, Mc);
    k_attnred<<<(int)((tmh + 255) / 256), 256, 0, stream>>>(A1, idxc, vf, P2,
                                                            res + (size_t)m0 * HID, Mc);
  }
  dim3 ga(cdiv(F, TBN), cdiv(N, TBM));
  k_gemm<<<ga, 256, 0, stream>>>(res, p.agg.w, p.agg.b, f_in, f_out, N, F, HID, 0);
}

static void run_td(const BNL* lyr, const float* l0pad, const float* xyz_in,
                   const float* f_in, int N, int S, int cin, int cout,
                   float* xyz_out, float* f_out, char* ws, hipStream_t stream)
{
  char* c = ws;
  int hw  = cin + 3;
  int hw4 = (hw + 3) & ~3;          // padded K
  int*   idx = (int*)c;   c += a256((size_t)S * K_NN * 4);
  float* h   = (float*)c; c += a256((size_t)S * K_NN * hw4 * 4);
  float* h1  = (float*)c; c += a256((size_t)S * K_NN * cout * 4);
  float* h2  = (float*)c; c += a256((size_t)S * K_NN * cout * 4);

  k_fps<<<1, 1024, 0, stream>>>(xyz_in, N, S, xyz_out);
  k_knn<<<cdiv(S, 256), 256, 0, stream>>>(xyz_out, S, xyz_in, N, idx);
  k_rel<<<cdiv(S * K_NN, 256), 256, 0, stream>>>(xyz_out, xyz_in, idx, h, S, hw4, 0, -1.0f);
  size_t tg = (size_t)S * K_NN * cin;
  k_gatherf<<<(int)((tg + 255) / 256), 256, 0, stream>>>(f_in, idx, h, S, cin, hw4, 3);

  dim3 g1(cdiv(cout, TBN), cdiv(S * K_NN, TBM));
  size_t tt = (size_t)S * K_NN * cout;
  k_gemm<<<g1, 256, 0, stream>>>(h, l0pad, lyr[0].b, nullptr, h1, S * K_NN, cout, hw4, 0);
  k_bnrelu<<<(int)((tt + 255) / 256), 256, 0, stream>>>(h1, lyr[0].mean, lyr[0].var,
                                                        lyr[0].gamma, lyr[0].beta, tt, cout);
  k_gemm<<<g1, 256, 0, stream>>>(h1, lyr[1].w, lyr[1].b, nullptr, h2, S * K_NN, cout, cout, 0);
  k_bnrelu<<<(int)((tt + 255) / 256), 256, 0, stream>>>(h2, lyr[1].mean, lyr[1].var,
                                                        lyr[1].gamma, lyr[1].beta, tt, cout);
  size_t tp = (size_t)S * cout;
  k_maxpool16<<<(int)((tp + 255) / 256), 256, 0, stream>>>(h2, f_out, S, cout);
}

extern "C" void kernel_launch(void* const* d_in, const int* in_sizes, int n_in,
                              void* d_out, int out_size, void* d_ws, size_t ws_size,
                              hipStream_t stream)
{
  (void)in_sizes; (void)n_in; (void)out_size; (void)ws_size;
  int pi = 0;
  auto nxt = [&]() { return (const float*)d_in[pi++]; };

  // pytree flatten order: events, then params dict sorted:
  // down < kernel_fn < kt < t1 < t2 ; inner dicts sorted by key.
  const float* events = nxt();
  BNL down[4][2];
  for (int i = 0; i < 4; ++i)
    for (int l = 0; l < 2; ++l) {
      down[i][l].b = nxt(); down[i][l].beta = nxt(); down[i][l].gamma = nxt();
      down[i][l].mean = nxt(); down[i][l].var = nxt(); down[i][l].w = nxt();
    }
  Lin kfn0, kfn1;
  kfn0.b = nxt(); kfn0.w = nxt();
  kfn1.b = nxt(); kfn1.w = nxt();
  auto parse_tb = [&]() {
    TBp t;
    t.agg.b  = nxt(); t.agg.w  = nxt();
    t.att0.b = nxt(); t.att0.w = nxt();
    t.att1.b = nxt(); t.att1.w = nxt();
    t.ker.b  = nxt(); t.ker.w  = nxt();
    t.pos0.b = nxt(); t.pos0.w = nxt();
    t.pos1.b = nxt(); t.pos1.w = nxt();
    t.wk = nxt(); t.wq = nxt(); t.wv = nxt();
    t.pos0w_pad = nullptr;
    return t;
  };
  TBp kt = parse_tb();
  TBp t1[4]; for (int i = 0; i < 4; ++i) t1[i] = parse_tb();
  TBp t2[3]; for (int i = 0; i < 3; ++i) t2[i] = parse_tb();

  // output layout: (xyz,f) x 5 levels, flat f32
  float* out  = (float*)d_out;
  float* xyz0 = out;
  float* f0   = xyz0 + 4096 * 3;
  float* xyz1 = f0   + 4096 * 32;
  float* f1   = xyz1 + 1024 * 3;
  float* xyz2 = f1   + 1024 * 64;
  float* f2   = xyz2 + 256 * 3;
  float* xyz3 = f2   + 256 * 128;
  float* f3   = xyz3 + 64 * 3;
  float* xyz4 = f3   + 64 * 256;
  float* f4   = xyz4 + 16 * 3;

  // workspace: persistent region (feature tmps + padded weights) + scratch
  char* wsp = (char*)d_ws;
  float* ftA = (float*)wsp; wsp += a256(4096 * 32 * 4);
  float* ftB = (float*)wsp; wsp += a256(4096 * 32 * 4);

  const int Ns[5] = {4096, 1024, 256, 64, 16};
  const int FD[5] = {32, 64, 128, 256, 512};

  // padded pos0 weights (3x512 -> 4x512) for all 8 transformer blocks
  TBp* tbs[8] = {&kt, &t1[0], &t1[1], &t1[2], &t1[3], &t2[0], &t2[1], &t2[2]};
  for (int i = 0; i < 8; ++i) {
    float* pp = (float*)wsp; wsp += a256(4 * HID * 4);
    k_padw<<<cdiv(4 * HID, 256), 256, 0, stream>>>(tbs[i]->pos0.w, pp, 3, 4, HID);
    tbs[i]->pos0w_pad = pp;
  }
  // padded TD layer-0 weights ((cin+3)xcout -> (cin+4)xcout)
  float* l0pad[4];
  for (int i = 0; i < 4; ++i) {
    int hw = FD[i] + 3, hw4 = (hw + 3) & ~3, cout = FD[i + 1];
    l0pad[i] = (float*)wsp; wsp += a256((size_t)hw4 * cout * 4);
    k_padw<<<cdiv(hw4 * cout, 256), 256, 0, stream>>>(down[i][0].w, l0pad[i], hw, hw4, cout);
  }
  char* scratch = wsp;

  // xyz0 = events[..., :3]
  k_slice3<<<cdiv(4096, 256), 256, 0, stream>>>(events, xyz0, 4096);

  // kernel_fn: 4 -> 32 (relu) -> 32
  dim3 gk(cdiv(32, TBN), cdiv(4096, TBM));
  k_gemm<<<gk, 256, 0, stream>>>(events, kfn0.w, kfn0.b, nullptr, ftA, 4096, 32, 4, 1);
  k_gemm<<<gk, 256, 0, stream>>>(ftA, kfn1.w, kfn1.b, nullptr, ftB, 4096, 32, 32, 0);

  // kt transformer block at N=4096, F=32
  run_tb(kt, xyz0, ftB, f0, 4096, 32, scratch, stream);

  float* xyzs[5] = {xyz0, xyz1, xyz2, xyz3, xyz4};
  float* fs[5]   = {f0, f1, f2, f3, f4};
  for (int i = 0; i < 4; ++i) {
    run_td(down[i], l0pad[i], xyzs[i], fs[i], Ns[i], Ns[i + 1], FD[i], FD[i + 1],
           xyzs[i + 1], fs[i + 1], scratch, stream);
    run_tb(t1[i], xyzs[i + 1], fs[i + 1], fs[i + 1], Ns[i + 1], FD[i + 1], scratch, stream);
    if (i < 3)
      run_tb(t2[i], xyzs[i + 1], fs[i + 1], fs[i + 1], Ns[i + 1], FD[i + 1], scratch, stream);
  }
}
